// Qwen3TTSAttention_69157563400871
// MI455X (gfx1250) — matrix operand order repaired and checked by reading, not compiled, not run on hardware
//
#include <hip/hip_runtime.h>
#include <hip/hip_bf16.h>

// ---------------------------------------------------------------------------
// Qwen3-TTS attention layer for MI455X (gfx1250), wave32 + WMMA bf16 + TDM.
//   B=1, S=4096, HID=2048, H=16, KV=8, D=128, GQA groups=2.
// Pipeline: [f32->bf16 convert] -> [fused QKV GEMM + RMSnorm + RoPE (+Vt)]
//           -> [flash attention: TDM double-buffered K/V in LDS, all-WMMA]
//           -> [O-proj GEMM, f32 out]
// ---------------------------------------------------------------------------

typedef __bf16 bf16_t;
typedef __attribute__((ext_vector_type(16))) __bf16 v16bf;
typedef __attribute__((ext_vector_type(8)))  float  v8f;
typedef __attribute__((ext_vector_type(4)))  unsigned int v4u;
typedef __attribute__((ext_vector_type(8)))  int v8i;
typedef __attribute__((ext_vector_type(4)))  int v4i;

static constexpr int S_   = 4096;
static constexpr int HID_ = 2048;
static constexpr int H_   = 16;
static constexpr int KV_  = 8;
static constexpr int D_   = 128;
static constexpr float EPS_     = 1e-6f;
static constexpr float SCALING_ = 0.08838834764831845f;  // D^-0.5

// --- WMMA fragment loaders (this file's contiguous k order k = 16*(lane/16) + s, same for A and B) ------
// A (16x32 bf16): lanes 0-15: m=lane, K=k0+0..15;
//                 lanes 16-31: m=lane-16, K=k0+16..31
__device__ __forceinline__ v16bf load_a_frag(const bf16_t* __restrict__ row,
                                             int k0, int lane) {
  const int off = (lane >> 4) ? 16 : 0;
  union { v16bf v; uint4 q[2]; } u;
  u.q[0] = *(const uint4*)(row + k0 + off);
  u.q[1] = *(const uint4*)(row + k0 + off + 8);
  return u.v;
}
// B (32x16 bf16), source arranged so K is contiguous per column:
//   lanes 0-15: n=lane, K=k0+0..15 ; lanes 16-31: n=lane-16, K=k0+16..31
__device__ __forceinline__ v16bf load_b_frag(const bf16_t* __restrict__ col,
                                             int k0, int lane) {
  const int off = (lane >> 4) ? 16 : 0;
  return *(const v16bf*)(col + k0 + off);
}
// B fragment from LDS with 16B (not 32B) row alignment: two b128 loads.
__device__ __forceinline__ v16bf load_b_frag_lds(const bf16_t* row, int lane) {
  const int off = (lane >> 4) ? 16 : 0;
  union { v16bf v; uint4 q[2]; } u;
  u.q[0] = *(const uint4*)(row + off);
  u.q[1] = *(const uint4*)(row + off + 8);
  return u.v;
}
// C/D (16x16 f32): vgpr r holds m = (lane/16)*8 + r, n = lane%16.

// ---------------------------------------------------------------------------
// TDM: 2D tile load global->LDS, D# packed per CDNA5 ISA
// (count=1/type=2 in group0; data_size=2B, pad_enable + pad codes, tensor/tile
// dims and dim0 stride in group1; groups 2/3 and the trailing group zero for a
// 2D tile). This toolchain exposes the 6-arg builtin (clang-23 / therock HIP).
// ---------------------------------------------------------------------------
__device__ __forceinline__ void tdm_load_2d(unsigned lds_off, const void* gptr,
                                            unsigned td0, unsigned td1,
                                            unsigned long long stride0,
                                            unsigned tile0, unsigned tile1,
                                            unsigned pad_int, unsigned pad_amt) {
  const unsigned long long ga = (unsigned long long)(uintptr_t)gptr;
  union { v4u v; unsigned u[4]; } g0;
  g0.u[0] = 1u;                                     // count=1, user-mode D#
  g0.u[1] = lds_off;                                // [63:32]  lds_addr
  g0.u[2] = (unsigned)ga;                           // [95:64]  global_addr lo
  g0.u[3] = (unsigned)((ga >> 32) & 0x01ffffffu)    // [120:96] global_addr hi
          | (2u << 30);                             // [127:126] type=2 (image)
  union { v8i v; unsigned long long q[4]; } g1;
  const unsigned long long flags =
      (1ull << 16)                                  // data_size = 1 -> 2 bytes
    | (1ull << 20)                                  // pad_enable
    | ((unsigned long long)pad_int << 22)           // pad_interval code
    | ((unsigned long long)pad_amt << 25);          // pad_amount code
  g1.q[0] = flags | ((unsigned long long)(td0 & 0xffffu) << 48);
  g1.q[1] = ((unsigned long long)td0 >> 16)
          | ((unsigned long long)td1 << 16)
          | ((unsigned long long)(tile0 & 0xffffu) << 48);
  g1.q[2] = (unsigned long long)(tile1 & 0xffffu)   // tile_dim1 ; tile_dim2=0
          | ((stride0 & 0xffffffffull) << 32);      // dim0_stride[31:0]
  g1.q[3] = ((stride0 >> 32) & 0xffffull);          // dim0_stride[47:32]; dim1_stride=0
  union { v4i v; unsigned u[4]; } gz;
  gz.u[0] = gz.u[1] = gz.u[2] = gz.u[3] = 0;
  union { v8i v; unsigned long long q[4]; } gz8;
  gz8.q[0] = gz8.q[1] = gz8.q[2] = gz8.q[3] = 0ull;
  __builtin_amdgcn_tensor_load_to_lds(g0.v, g1.v, gz.v, gz.v, gz8.v, 0);
}

// ---------------------------------------------------------------------------
// Stage 0: fp32 -> bf16 conversion (grid-stride)
// ---------------------------------------------------------------------------
__global__ void convert_f32_bf16(const float* __restrict__ src,
                                 bf16_t* __restrict__ dst, int n) {
  int i = blockIdx.x * blockDim.x + threadIdx.x;
  const int stride = gridDim.x * blockDim.x;
  for (; i < n; i += stride) dst[i] = (bf16_t)src[i];
}

// ---------------------------------------------------------------------------
// Stage 1: fused QKV projection + RMSnorm + RoPE epilogue (+V transpose).
// Block = 256 threads = 8 waves (4 M x 2 N); tile 128x128; K-step 32.
// ---------------------------------------------------------------------------
__global__ __launch_bounds__(256)
void qkv_gemm_kernel(const bf16_t* __restrict__ hid,
                     const bf16_t* __restrict__ wqkv,
                     const float* __restrict__ cosp,
                     const float* __restrict__ sinp,
                     const float* __restrict__ qnw,
                     const float* __restrict__ knw,
                     bf16_t* __restrict__ qout,   // [H][S][D]
                     bf16_t* __restrict__ kout,   // [KV][S][D]
                     bf16_t* __restrict__ vt)     // [KV][D][S]
{
  __shared__ float  rowsq[128];
  __shared__ bf16_t tile[128][130];

  const int tid  = threadIdx.x;
  const int lane = tid & 31;
  const int wv_  = tid >> 5;
  const int wm   = wv_ & 3, wn = wv_ >> 2;
  const int hf   = lane >> 4, ln = lane & 15;
  const int m0   = blockIdx.x * 128;
  const int nb   = blockIdx.y;
  const int n0   = nb * 128;

  v8f acc[2][4] = {};

  for (int k0 = 0; k0 < HID_; k0 += 32) {
    v16bf a[2], b[4];
#pragma unroll
    for (int mt = 0; mt < 2; ++mt)
      a[mt] = load_a_frag(hid + (size_t)(m0 + wm * 32 + mt * 16 + ln) * HID_, k0, lane);
#pragma unroll
    for (int nt = 0; nt < 4; ++nt)
      b[nt] = load_b_frag(wqkv + (size_t)(n0 + wn * 64 + nt * 16 + ln) * HID_, k0, lane);
#pragma unroll
    for (int mt = 0; mt < 2; ++mt)
#pragma unroll
      for (int nt = 0; nt < 4; ++nt)
        acc[mt][nt] = __builtin_amdgcn_wmma_f32_16x16x32_bf16(
            false, a[mt], false, b[nt], (short)0, acc[mt][nt], false, false);
  }

  if (tid < 128) rowsq[tid] = 0.0f;
  __syncthreads();
  if (nb < 24) {
#pragma unroll
    for (int mt = 0; mt < 2; ++mt) {
#pragma unroll
      for (int r = 0; r < 8; ++r) {
        float ss = 0.0f;
#pragma unroll
        for (int nt = 0; nt < 4; ++nt) { float v = acc[mt][nt][r]; ss += v * v; }
        for (int mk = 1; mk < 16; mk <<= 1) ss += __shfl_xor(ss, mk, 16);
        if (ln == 0) atomicAdd(&rowsq[wm * 32 + mt * 16 + hf * 8 + r], ss);
      }
    }
  }
  __syncthreads();

  const float* nw = (nb < 16) ? qnw : knw;
#pragma unroll
  for (int mt = 0; mt < 2; ++mt) {
    const int rbase = wm * 32 + mt * 16 + hf * 8;
#pragma unroll
    for (int r = 0; r < 8; ++r) {
      const int row = rbase + r;
      float rstd = 1.0f;
      if (nb < 24) rstd = rsqrtf(rowsq[row] * (1.0f / 128.0f) + EPS_);
#pragma unroll
      for (int nt = 0; nt < 4; ++nt) {
        const int col = wn * 64 + nt * 16 + ln;
        float v = acc[mt][nt][r];
        if (nb < 24) v = v * rstd * nw[col];
        tile[row][col] = (bf16_t)v;
      }
    }
  }
  __syncthreads();

  if (nb < 16) {            // Q: RoPE + 1/sqrt(D)
    const int h = nb;
    for (int idx = tid; idx < 128 * 128; idx += 256) {
      const int row = idx >> 7, d = idx & 127;
      const int s = m0 + row;
      const float x  = (float)tile[row][d];
      const float xp = (float)tile[row][d ^ 64];
      const float c  = cosp[s * D_ + d], sn = sinp[s * D_ + d];
      const float rot = (d < 64) ? -xp : xp;
      qout[((size_t)h * S_ + s) * D_ + d] = (bf16_t)((x * c + rot * sn) * SCALING_);
    }
  } else if (nb < 24) {     // K: RoPE
    const int h = nb - 16;
    for (int idx = tid; idx < 128 * 128; idx += 256) {
      const int row = idx >> 7, d = idx & 127;
      const int s = m0 + row;
      const float x  = (float)tile[row][d];
      const float xp = (float)tile[row][d ^ 64];
      const float c  = cosp[s * D_ + d], sn = sinp[s * D_ + d];
      const float rot = (d < 64) ? -xp : xp;
      kout[((size_t)h * S_ + s) * D_ + d] = (bf16_t)(x * c + rot * sn);
    }
  } else {                  // V: transposed store vt[kv][d][s]
    const int h = nb - 24;
    for (int idx = tid; idx < 128 * 128; idx += 256) {
      const int d = idx >> 7, row = idx & 127;
      vt[((size_t)h * D_ + d) * S_ + (m0 + row)] = tile[row][d];
    }
  }
}

// ---------------------------------------------------------------------------
// Stage 2: causal flash attention, TDM double-buffered K/V in LDS.
// grid = (S/128, H); block = 256 = 8 waves; each wave owns 16 query rows.
// K chunk (32 keys x 128 d) and V chunk (128 d x 32 keys, from Vt) are DMA'd
// by wave 0 via tensor_load_to_lds with row padding (272B / 80B pitches:
// 16B-aligned ds_load_b128, dword strides 68/20 -> conflict-free mod 64).
// ---------------------------------------------------------------------------
__global__ __launch_bounds__(256)
void attn_kernel(const bf16_t* __restrict__ qin,   // [H][S][D]
                 const bf16_t* __restrict__ kin,   // [KV][S][D]
                 const bf16_t* __restrict__ vt,    // [KV][D][S]
                 bf16_t* __restrict__ aout)        // [S][H*D]
{
  constexpr int KPITCH = 136;   // bf16 elems: 256B row + 4-dword TDM pad
  constexpr int VPITCH = 40;    // bf16 elems:  64B row + 4-dword TDM pad
  __shared__ __attribute__((aligned(16))) bf16_t kstage[2][32 * KPITCH];
  __shared__ __attribute__((aligned(16))) bf16_t vstage[2][128 * VPITCH];
  __shared__ __attribute__((aligned(16))) bf16_t pstage[8][16][48];

  const int tid  = threadIdx.x;
  const int lane = tid & 31;
  const int wv_  = tid >> 5;
  const int hf   = lane >> 4, ln = lane & 15;
  const int h    = blockIdx.y;
  const int hk   = h >> 1;                       // GQA groups = 2
  const int m0   = blockIdx.x * 128;
  const int q0   = m0 + wv_ * 16;

  // Q fragments: 4 x (16x32) over d, kept in registers
  v16bf qa[4];
  const bf16_t* qrow = qin + ((size_t)h * S_ + (q0 + ln)) * D_;
#pragma unroll
  for (int c = 0; c < 4; ++c) qa[c] = load_a_frag(qrow, c * 32, lane);

  v8f oacc[8] = {};
  float rmax[8], rsum[8];
#pragma unroll
  for (int r = 0; r < 8; ++r) { rmax[r] = -1e30f; rsum[r] = 0.0f; }

  const unsigned klds[2] = { (unsigned)(uintptr_t)&kstage[0][0],
                             (unsigned)(uintptr_t)&kstage[1][0] };
  const unsigned vlds[2] = { (unsigned)(uintptr_t)&vstage[0][0],
                             (unsigned)(uintptr_t)&vstage[1][0] };
  const bf16_t* kbase = kin + (size_t)hk * S_ * D_;
  const bf16_t* vbase = vt + (size_t)hk * D_ * S_;

  const int nchunks = m0 / 32 + 4;               // keys [0, m0+128)

  if (wv_ == 0) {  // prologue DMA: chunk 0 -> buffer 0
    tdm_load_2d(klds[0], kbase,      128, 32, 128u, 128, 32, /*pad*/5, 3);
    tdm_load_2d(vlds[0], vbase,       32, 128, (unsigned long long)S_, 32, 128, 3, 3);
  }

  for (int ci = 0; ci < nchunks; ++ci) {
    const int kc  = ci * 32;
    const int cur = ci & 1;
    if (wv_ == 0 && ci + 1 < nchunks) {          // DMA next chunk -> other buf
      const int nb = cur ^ 1, kn = kc + 32;
      tdm_load_2d(klds[nb], kbase + (size_t)kn * D_, 128, 32, 128u, 128, 32, 5, 3);
      tdm_load_2d(vlds[nb], vbase + kn, 32, 128, (unsigned long long)S_, 32, 128, 3, 3);
    }
    // in-order TENSORcnt: <=2 outstanding leaves only next chunk's pair in flight
    if (ci + 1 < nchunks) __builtin_amdgcn_s_wait_tensorcnt((short)2);
    else                  __builtin_amdgcn_s_wait_tensorcnt((short)0);
    __syncthreads();

    if (kc < q0 + 16) {                          // wave-uniform causal guard
      const bf16_t* kb_ = &kstage[cur][0];
      const bf16_t* vb_ = &vstage[cur][0];

      // ---- scores: two 16-key tiles, 4 WMMA K-steps over d each ----
      v8f sc[2] = {};
#pragma unroll
      for (int kt = 0; kt < 2; ++kt) {
        const bf16_t* krow = kb_ + (kt * 16 + ln) * KPITCH;
#pragma unroll
        for (int c = 0; c < 4; ++c) {
          v16bf kb = load_b_frag_lds(krow + c * 32, lane);
          sc[kt] = __builtin_amdgcn_wmma_f32_16x16x32_bf16(
              false, qa[c], false, kb, (short)0, sc[kt], false, false);
        }
      }

      // ---- online softmax on C-layout (row m = hf*8+r, key n = ln) ----
      float p0[8], p1[8];
#pragma unroll
      for (int r = 0; r < 8; ++r) {
        const int qpos = q0 + hf * 8 + r;
        float s0 = sc[0][r], s1 = sc[1][r];
        if (kc + ln > qpos)      s0 = -1e30f;
        if (kc + 16 + ln > qpos) s1 = -1e30f;
        float mx = fmaxf(s0, s1);
        for (int mk = 1; mk < 16; mk <<= 1) mx = fmaxf(mx, __shfl_xor(mx, mk, 16));
        mx = fmaxf(mx, rmax[r]);
        const float corr = __expf(rmax[r] - mx);
        rmax[r] = mx;
        const float e0 = __expf(s0 - mx), e1 = __expf(s1 - mx);
        float ps = e0 + e1;
        for (int mk = 1; mk < 16; mk <<= 1) ps += __shfl_xor(ps, mk, 16);
        rsum[r] = rsum[r] * corr + ps;
        p0[r] = e0; p1[r] = e1;
#pragma unroll
        for (int t = 0; t < 8; ++t) oacc[t][r] *= corr;
      }

      // ---- P: C-layout -> A-fragment via per-wave LDS scratch ----
#pragma unroll
      for (int r = 0; r < 8; ++r) {
        pstage[wv_][hf * 8 + r][ln]      = (bf16_t)p0[r];
        pstage[wv_][hf * 8 + r][16 + ln] = (bf16_t)p1[r];
      }
      asm volatile("s_wait_dscnt 0" ::: "memory");  // DS is wave-in-order
      v16bf pa;
      {
        union { v16bf v; uint4 q[2]; } u;
        const bf16_t* prow = &pstage[wv_][ln][0];
        const int off = hf ? 16 : 0;
        u.q[0] = *(const uint4*)(prow + off);
        u.q[1] = *(const uint4*)(prow + off + 8);
        pa = u.v;
      }

      // ---- P . V : 8 output d-tiles, K = 32 keys ----
#pragma unroll
      for (int t = 0; t < 8; ++t) {
        v16bf vb = load_b_frag_lds(vb_ + (t * 16 + ln) * VPITCH, lane);
        oacc[t] = __builtin_amdgcn_wmma_f32_16x16x32_bf16(
            false, pa, false, vb, (short)0, oacc[t], false, false);
      }
    }
    __syncthreads();                              // protect buffer being refilled
  }

  // ---- finalize: out = oacc / rsum ----
#pragma unroll
  for (int r = 0; r < 8; ++r) {
    const int s = q0 + hf * 8 + r;
    const float inv = 1.0f / rsum[r];
#pragma unroll
    for (int t = 0; t < 8; ++t)
      aout[(size_t)s * (H_ * D_) + h * D_ + t * 16 + ln] = (bf16_t)(oacc[t][r] * inv);
  }
}

// ---------------------------------------------------------------------------
// Stage 3: output projection, f32 result straight to d_out.
// ---------------------------------------------------------------------------
__global__ __launch_bounds__(256)
void oproj_kernel(const bf16_t* __restrict__ ain,  // [S][H*D]
                  const bf16_t* __restrict__ wo,   // [HID][H*D] bf16
                  float* __restrict__ out)         // [S][HID] f32
{
  const int tid  = threadIdx.x;
  const int lane = tid & 31;
  const int wv_  = tid >> 5;
  const int wm   = wv_ & 3, wn = wv_ >> 2;
  const int hf   = lane >> 4, ln = lane & 15;
  const int m0   = blockIdx.x * 128, n0 = blockIdx.y * 128;

  v8f acc[2][4] = {};
  for (int k0 = 0; k0 < H_ * D_; k0 += 32) {
    v16bf a[2], b[4];
#pragma unroll
    for (int mt = 0; mt < 2; ++mt)
      a[mt] = load_a_frag(ain + (size_t)(m0 + wm * 32 + mt * 16 + ln) * (H_ * D_), k0, lane);
#pragma unroll
    for (int nt = 0; nt < 4; ++nt)
      b[nt] = load_b_frag(wo + (size_t)(n0 + wn * 64 + nt * 16 + ln) * (H_ * D_), k0, lane);
#pragma unroll
    for (int mt = 0; mt < 2; ++mt)
#pragma unroll
      for (int nt = 0; nt < 4; ++nt)
        acc[mt][nt] = __builtin_amdgcn_wmma_f32_16x16x32_bf16(
            false, a[mt], false, b[nt], (short)0, acc[mt][nt], false, false);
  }
#pragma unroll
  for (int mt = 0; mt < 2; ++mt) {
#pragma unroll
    for (int r = 0; r < 8; ++r) {
      const int row = m0 + wm * 32 + mt * 16 + hf * 8 + r;
#pragma unroll
      for (int nt = 0; nt < 4; ++nt) {
        const int col = n0 + wn * 64 + nt * 16 + ln;
        out[(size_t)row * HID_ + col] = acc[mt][nt][r];
      }
    }
  }
}

// ---------------------------------------------------------------------------
extern "C" void kernel_launch(void* const* d_in, const int* in_sizes, int n_in,
                              void* d_out, int out_size, void* d_ws, size_t ws_size,
                              hipStream_t stream) {
  (void)in_sizes; (void)n_in; (void)out_size; (void)ws_size;
  const float* hidden = (const float*)d_in[0];
  const float* cosp   = (const float*)d_in[1];
  const float* sinp   = (const float*)d_in[2];
  const float* wq     = (const float*)d_in[3];
  const float* wk     = (const float*)d_in[4];
  const float* wv     = (const float*)d_in[5];
  const float* wo     = (const float*)d_in[6];
  const float* qnw    = (const float*)d_in[7];
  const float* knw    = (const float*)d_in[8];
  float* outp = (float*)d_out;

  bf16_t* p = (bf16_t*)d_ws;
  bf16_t* hidb  = p; p += (size_t)S_ * HID_;            // 16 MB
  bf16_t* wqkvb = p; p += (size_t)4096 * HID_;          // 16 MB (wq|wk|wv rows)
  bf16_t* wob   = p; p += (size_t)HID_ * (H_ * D_);     //  8 MB
  bf16_t* qb    = p; p += (size_t)H_ * S_ * D_;         // 16 MB
  bf16_t* kb    = p; p += (size_t)KV_ * S_ * D_;        //  8 MB
  bf16_t* vtb   = p; p += (size_t)KV_ * D_ * S_;        //  8 MB
  bf16_t* attnb = p; p += (size_t)S_ * (H_ * D_);       // 16 MB

  const dim3 cblk(256), cgrd(2048);
  convert_f32_bf16<<<cgrd, cblk, 0, stream>>>(hidden, hidb, S_ * HID_);
  convert_f32_bf16<<<cgrd, cblk, 0, stream>>>(wq, wqkvb,                 2048 * HID_);
  convert_f32_bf16<<<cgrd, cblk, 0, stream>>>(wk, wqkvb + 2048 * HID_,   1024 * HID_);
  convert_f32_bf16<<<cgrd, cblk, 0, stream>>>(wv, wqkvb + 3072 * HID_,   1024 * HID_);
  convert_f32_bf16<<<cgrd, cblk, 0, stream>>>(wo, wob,                   HID_ * (H_ * D_));

  qkv_gemm_kernel<<<dim3(S_ / 128, 32), 256, 0, stream>>>(
      hidb, wqkvb, cosp, sinp, qnw, knw, qb, kb, vtb);

  attn_kernel<<<dim3(S_ / 128, H_), 256, 0, stream>>>(qb, kb, vtb, attnb);

  oproj_kernel<<<dim3(S_ / 128, HID_ / 128), 256, 0, stream>>>(attnb, wob, outp);
}
